// TriangleMultiplicationOutgoing_19439021981757
// MI455X (gfx1250) — matrix operand order repaired and checked by reading, not compiled, not run on hardware
//
#include <hip/hip_runtime.h>
#include <math.h>

typedef __attribute__((ext_vector_type(16))) __bf16        v16bf;
typedef __attribute__((ext_vector_type(8)))  float         v8f;
typedef __attribute__((ext_vector_type(4)))  float         v4f;
typedef __attribute__((ext_vector_type(4)))  unsigned int  v4u;

// GCC-vector int4 flavors for the async-copy builtin's exact prototype
typedef int vs4i __attribute__((vector_size(16)));
typedef __attribute__((address_space(1))) vs4i as1_vs4i;
typedef __attribute__((address_space(3))) vs4i as3_vs4i;

#define NSEQ 512
#define DCH  128
#define N2   (NSEQ * NSEQ)     // 262144: elements of one per-channel matrix
#define MROWS (NSEQ * NSEQ)    // 262144: flattened (i,j) rows

// ---- fast math (hardware rcp/rsq instead of IEEE divide expansions) --------
__device__ __forceinline__ float fast_rcp(float x) {
#if __has_builtin(__builtin_amdgcn_rcpf)
  return __builtin_amdgcn_rcpf(x);
#else
  return 1.0f / x;
#endif
}
__device__ __forceinline__ float fast_rsqrt(float x) {
#if __has_builtin(__builtin_amdgcn_rsqf)
  return __builtin_amdgcn_rsqf(x);
#else
  return rsqrtf(x);
#endif
}
__device__ __forceinline__ float sigmoidf_(float x) { return fast_rcp(1.0f + __expf(-x)); }

// ---- CDNA5 async global->LDS copies (ASYNCcnt), with safe fallback ---------
__device__ __forceinline__ void async_copy16(const void* g, void* l) {
#if __has_builtin(__builtin_amdgcn_global_load_async_to_lds_b128)
  __builtin_amdgcn_global_load_async_to_lds_b128(
      (as1_vs4i*)g, (as3_vs4i*)l, 0, 0);
#else
  *(v4u*)l = *(const v4u*)g;
#endif
}
__device__ __forceinline__ void wait_async0() {
#if __has_builtin(__builtin_amdgcn_s_wait_asynccnt)
  __builtin_amdgcn_s_wait_asynccnt(0);
#elif defined(__gfx1250__)
  asm volatile("s_wait_asynccnt 0" ::: "memory");
#endif
}

// ---- CDNA5 WMMA bf16 fragment loaders (wave32) -----------------------------
// A-matrix 16x32 bf16: lane L holds row m = L%16.
//   lanes 0-15 : K = k0+0..7 then k0+16..23
//   lanes 16-31: K = k0+8..15 then k0+24..31
__device__ __forceinline__ v16bf load_frag_a(const __bf16* A, int ldm, int m0, int k0, int lane) {
  const __bf16* p = A + (size_t)(m0 + (lane & 15)) * ldm + k0 + ((lane >> 4) << 3);
  union { v16bf v; v4u q[2]; } f;
  f.q[0] = *(const v4u*)(p);
  f.q[1] = *(const v4u*)(p + 16);
  return f.v;
}
// B-matrix 32x16 bf16 (stored as [n][k] row-major, B[k][n] = W[n][k]):
//   lane L holds col n = L%16; lanes 0-15: K = k0+0..7 then k0+16..23; lanes 16-31: K = k0+8..15 then k0+24..31
__device__ __forceinline__ v16bf load_frag_b(const __bf16* W, int ldn, int n0, int k0, int lane) {
  const __bf16* p = W + (size_t)(n0 + (lane & 15)) * ldn + k0 + ((lane >> 4) << 3);
  union { v16bf v; v4u q[2]; } f;
  f.q[0] = *(const v4u*)(p);
  f.q[1] = *(const v4u*)(p + 16);
  return f.v;
}

__device__ __forceinline__ v8f wmma_bf16(v16bf a, v16bf b, v8f c) {
  return __builtin_amdgcn_wmma_f32_16x16x32_bf16(false, a, false, b, (short)0, c, false, false);
}

// ---- Kernel 0: convert all weights f32 -> bf16 once ------------------------
// wbf layout (elements): p_in[0..32767], g_in[32768..65535],
//                        g_out[65536..81919], p_out[81920..98303]
__global__ __launch_bounds__(256)
void prep_weights_kernel(const float* __restrict__ pin, const float* __restrict__ gin,
                         const float* __restrict__ gow, const float* __restrict__ pot,
                         __bf16* __restrict__ wbf) {
  int idx = blockIdx.x * 256 + threadIdx.x;
  if (idx < 32768)       wbf[idx] = (__bf16)pin[idx];
  else if (idx < 65536)  wbf[idx] = (__bf16)gin[idx - 32768];
  else if (idx < 81920)  wbf[idx] = (__bf16)gow[idx - 65536];
  else if (idx < 98304)  wbf[idx] = (__bf16)pot[idx - 81920];
}

// ---- Kernel 1: fused LayerNorm + p_in/g_in/g_out projections ---------------
// 64 rows per block. Writes:
//   a_ws[d][i][k], b_ws[d][i][k] (bf16, channel-major for the einsum)
//   gate_ws[c][m]               (bf16, sigmoid(x_in @ g_out_w.T), transposed)
__global__ __launch_bounds__(256)
void fused_ln_proj_kernel(const float* __restrict__ x, const float* __restrict__ mask,
                          const float* __restrict__ nw, const float* __restrict__ nb,
                          const __bf16* __restrict__ wbf,
                          __bf16* __restrict__ a_ws, __bf16* __restrict__ b_ws,
                          __bf16* __restrict__ gate_ws) {
  __shared__ float  xs[64][132];
  __shared__ __bf16 xin[64][136];
  const int tid  = threadIdx.x;
  const int lane = tid & 31;
  const int wave = tid >> 5;
  const int gm0  = blockIdx.x * 64;

  // async copy 64x128 f32 tile into LDS (rows padded to 132)
  for (int e = tid * 4; e < 64 * 128; e += 1024) {
    int r = e >> 7, c = e & 127;
    async_copy16(x + (size_t)(gm0 + r) * 128 + c, &xs[r][c]);
  }
  wait_async0();
  __syncthreads();

  { // LayerNorm: 4 lanes per row, 32 elems each
    int r = tid >> 2, c0 = (tid & 3) * 32;
    float s = 0.f;
    #pragma unroll
    for (int e = 0; e < 32; ++e) s += xs[r][c0 + e];
    s += __shfl_xor(s, 1); s += __shfl_xor(s, 2);
    float mu = s * (1.0f / 128.0f);
    float v = 0.f;
    #pragma unroll
    for (int e = 0; e < 32; ++e) { float d = xs[r][c0 + e] - mu; v += d * d; }
    v += __shfl_xor(v, 1); v += __shfl_xor(v, 2);
    float rs = fast_rsqrt(v * (1.0f / 128.0f) + 1e-5f);
    #pragma unroll
    for (int e = 0; e < 32; ++e) {
      int c = c0 + e;
      xin[r][c] = (__bf16)((xs[r][c] - mu) * rs * nw[c] + nb[c]);
    }
  }
  __syncthreads();

  const __bf16* pw  = wbf;          // [256][128]
  const __bf16* gw  = wbf + 32768;  // [256][128]
  const __bf16* gow = wbf + 65536;  // [128][128]
  const int hi = lane >> 4, nn = lane & 15;

  for (int job = wave; job < 96; job += 8) {
    if (job < 64) {                       // paired p_in / g_in tile (m,n)
      int m = job & 3, n = job >> 2;      // n: 0..15 over 256 cols
      v8f accP = {}; v8f accG = {};
      #pragma unroll
      for (int k0 = 0; k0 < 128; k0 += 32) {
        v16bf a  = load_frag_a(&xin[0][0], 136, m * 16, k0, lane);
        v16bf bp = load_frag_b(pw, 128, n * 16, k0, lane);
        accP = wmma_bf16(a, bp, accP);
        v16bf bg = load_frag_b(gw, 128, n * 16, k0, lane);
        accG = wmma_bf16(a, bg, accG);
      }
      int c = n * 16 + nn;
      int gmB = gm0 + m * 16 + hi * 8;    // 8 consecutive flattened rows
      int i = gmB >> 9, kb = gmB & 511;
      const float* mrow = mask + (size_t)i * 512 + kb;
      __align__(16) __bf16 outv[8];
      #pragma unroll
      for (int r = 0; r < 8; ++r)
        outv[r] = (__bf16)(accP[r] * sigmoidf_(accG[r]) * mrow[r]);
      __bf16* dst = (c < 128) ? (a_ws + (size_t)c * N2 + (size_t)i * 512 + kb)
                              : (b_ws + (size_t)(c - 128) * N2 + (size_t)i * 512 + kb);
      *(v4u*)dst = *(const v4u*)outv;     // transposed 16B store
    } else {                              // g_out gate tile
      int j2 = job - 64;
      int m = j2 & 3, n = j2 >> 2;        // n: 0..7 over 128 cols
      v8f acc = {};
      #pragma unroll
      for (int k0 = 0; k0 < 128; k0 += 32) {
        v16bf a  = load_frag_a(&xin[0][0], 136, m * 16, k0, lane);
        v16bf bg = load_frag_b(gow, 128, n * 16, k0, lane);
        acc = wmma_bf16(a, bg, acc);
      }
      int c = n * 16 + nn;
      int gmB = gm0 + m * 16 + hi * 8;
      __align__(16) __bf16 outv[8];
      #pragma unroll
      for (int r = 0; r < 8; ++r) outv[r] = (__bf16)sigmoidf_(acc[r]);
      *(v4u*)(gate_ws + (size_t)c * MROWS + gmB) = *(const v4u*)outv;
    }
  }
}

// ---- Kernel 2: t[d][i][j] = sum_k a[d][i][k] * b[d][j][k] ------------------
// Per block: one channel d, one 128(i) x 128(j) tile, computed as T^T
// (A-operand = b rows j, B-operand = a rows i) so stores are contiguous in j.
// Double-buffered LDS staging via GLOBAL_LOAD_ASYNC_TO_LDS (ASYNCcnt).
__global__ __launch_bounds__(256)
void triangle_einsum_kernel(const __bf16* __restrict__ a_ws, const __bf16* __restrict__ b_ws,
                            float* __restrict__ t_ws) {
  __shared__ __bf16 Ajs[2][128][40];  // rows j from b_ws, 32 k-elems (+pad)
  __shared__ __bf16 Bis[2][128][40];  // rows i from a_ws
  const int tid  = threadIdx.x;
  const int lane = tid & 31;
  const int wave = tid >> 5;
  const int d  = blockIdx.x >> 4;
  const int tl = blockIdx.x & 15;
  const int i0 = (tl >> 2) * 128;
  const int j0 = (tl & 3) * 128;
  const int wm = wave & 3;         // 4 waves along M (= j)
  const int wn = wave >> 2;        // 2 waves along N (= i)
  const size_t dOff = (size_t)d * N2;
  const __bf16* aBase = a_ws + dOff + (size_t)i0 * 512;
  const __bf16* bBase = b_ws + dOff + (size_t)j0 * 512;

  // issue one 32-wide k-slab of both operand tiles into buffer `bsel`
  auto issue_slab = [&](int bsel, int kk) {
    #pragma unroll
    for (int t = tid; t < 512; t += 256) {     // 128 rows x 4 chunks of 16B
      int row = t >> 2, ch = (t & 3) * 8;
      async_copy16(bBase + (size_t)row * 512 + kk + ch, &Ajs[bsel][row][ch]);
      async_copy16(aBase + (size_t)row * 512 + kk + ch, &Bis[bsel][row][ch]);
    }
  };

  v8f acc[2][4] = {};
  int bsel = 0;
  issue_slab(0, 0);
  for (int kk = 0; kk < 512; kk += 32) {
    wait_async0();           // this wave's async writes to LDS are complete
    __syncthreads();         // everyone's writes complete; prior compute done
    if (kk + 32 < 512) issue_slab(bsel ^ 1, kk + 32);
    v16bf af[2];
    #pragma unroll
    for (int mt = 0; mt < 2; ++mt)
      af[mt] = load_frag_a(&Ajs[bsel][0][0], 40, wm * 32 + mt * 16, 0, lane);
    #pragma unroll
    for (int nt = 0; nt < 4; ++nt) {
      v16bf bfr = load_frag_b(&Bis[bsel][0][0], 40, wn * 64 + nt * 16, 0, lane);
      #pragma unroll
      for (int mt = 0; mt < 2; ++mt)
        acc[mt][nt] = wmma_bf16(af[mt], bfr, acc[mt][nt]);
    }
    bsel ^= 1;
  }

  const int hi = lane >> 4, nn = lane & 15;
  #pragma unroll
  for (int mt = 0; mt < 2; ++mt) {
    #pragma unroll
    for (int nt = 0; nt < 4; ++nt) {
      int ig = i0 + wn * 64 + nt * 16 + nn;       // N coord = i (fixed per lane)
      int jb = j0 + wm * 32 + mt * 16 + hi * 8;   // M coords = 8 consecutive j
      float* dst = t_ws + dOff + (size_t)ig * 512 + jb;
      v4f lo = { acc[mt][nt][0], acc[mt][nt][1], acc[mt][nt][2], acc[mt][nt][3] };
      v4f hh = { acc[mt][nt][4], acc[mt][nt][5], acc[mt][nt][6], acc[mt][nt][7] };
      *(v4f*)dst       = lo;
      *(v4f*)(dst + 4) = hh;
    }
  }
}

// ---- Kernel 3: out = LN(t) @ p_out_w.T * gate ------------------------------
__global__ __launch_bounds__(256)
void ln_out_proj_kernel(const float* __restrict__ t_ws, const __bf16* __restrict__ powb,
                        const __bf16* __restrict__ gate_ws,
                        const float* __restrict__ nw, const float* __restrict__ nb,
                        float* __restrict__ out) {
  __shared__ float  ts[64][132];
  __shared__ __bf16 tn[64][136];
  const int tid  = threadIdx.x;
  const int lane = tid & 31;
  const int wave = tid >> 5;
  const int gm0  = blockIdx.x * 64;
  const int i  = gm0 >> 9;
  const int j0 = gm0 & 511;

  // gather t[d][i][j0..j0+63] for all d  ->  ts[j][d]  (transpose in LDS)
  for (int e = tid * 4; e < 8192; e += 1024) {
    int dd = e >> 6, jj = e & 63;
    v4f v = *(const v4f*)(t_ws + (size_t)dd * N2 + (size_t)i * 512 + j0 + jj);
    ts[jj + 0][dd] = v[0]; ts[jj + 1][dd] = v[1];
    ts[jj + 2][dd] = v[2]; ts[jj + 3][dd] = v[3];
  }
  __syncthreads();

  { // LayerNorm over channels
    int r = tid >> 2, c0 = (tid & 3) * 32;
    float s = 0.f;
    #pragma unroll
    for (int e = 0; e < 32; ++e) s += ts[r][c0 + e];
    s += __shfl_xor(s, 1); s += __shfl_xor(s, 2);
    float mu = s * (1.0f / 128.0f);
    float v = 0.f;
    #pragma unroll
    for (int e = 0; e < 32; ++e) { float d = ts[r][c0 + e] - mu; v += d * d; }
    v += __shfl_xor(v, 1); v += __shfl_xor(v, 2);
    float rs = fast_rsqrt(v * (1.0f / 128.0f) + 1e-5f);
    #pragma unroll
    for (int e = 0; e < 32; ++e) {
      int c = c0 + e;
      tn[r][c] = (__bf16)((ts[r][c] - mu) * rs * nw[c] + nb[c]);
    }
  }
  __syncthreads();

  const int hi = lane >> 4, nn = lane & 15;
  for (int job = wave; job < 32; job += 8) {
    int m = job & 3, n = job >> 2;
    v8f acc = {};
    #pragma unroll
    for (int k0 = 0; k0 < 128; k0 += 32) {
      v16bf a = load_frag_a(&tn[0][0], 136, m * 16, k0, lane);
      v16bf b = load_frag_b(powb, 128, n * 16, k0, lane);
      acc = wmma_bf16(a, b, acc);
    }
    int c  = n * 16 + nn;
    int mB = m * 16 + hi * 8;
    union { v4u q; __bf16 h[8]; } g;
    g.q = *(const v4u*)(gate_ws + (size_t)c * MROWS + gm0 + mB);
    #pragma unroll
    for (int r = 0; r < 8; ++r) ts[mB + r][c] = acc[r] * (float)g.h[r];  // stage via LDS
  }
  __syncthreads();

  for (int e = tid * 4; e < 8192; e += 1024) {   // coalesced f32 output
    int row = e >> 7, col = e & 127;
    v4f v = { ts[row][col], ts[row][col + 1], ts[row][col + 2], ts[row][col + 3] };
    *(v4f*)(out + (size_t)(gm0 + row) * 128 + col) = v;
  }
}

extern "C" void kernel_launch(void* const* d_in, const int* in_sizes, int n_in,
                              void* d_out, int out_size, void* d_ws, size_t ws_size,
                              hipStream_t stream) {
  const float* x    = (const float*)d_in[0];
  const float* mask = (const float*)d_in[1];
  const float* niw  = (const float*)d_in[2];
  const float* nib  = (const float*)d_in[3];
  const float* pin  = (const float*)d_in[4];
  const float* gin  = (const float*)d_in[5];
  const float* now  = (const float*)d_in[6];
  const float* nob  = (const float*)d_in[7];
  const float* pot  = (const float*)d_in[8];
  const float* gow  = (const float*)d_in[9];
  float* out = (float*)d_out;

  char* ws = (char*)d_ws;
  __bf16* a_ws    = (__bf16*)(ws);                     //  64 MB
  __bf16* b_ws    = (__bf16*)(ws + 67108864ull);       //  64 MB
  __bf16* gate_ws = (__bf16*)(ws + 134217728ull);      //  64 MB
  float*  t_ws    = (float*) (ws + 201326592ull);      // 128 MB
  __bf16* wbf     = (__bf16*)(ws + 335544320ull);      // 192 KB

  prep_weights_kernel<<<384, 256, 0, stream>>>(pin, gin, gow, pot, wbf);
  fused_ln_proj_kernel<<<4096, 256, 0, stream>>>(x, mask, niw, nib, wbf, a_ws, b_ws, gate_ws);
  triangle_einsum_kernel<<<2048, 256, 0, stream>>>(a_ws, b_ws, t_ws);
  ln_out_proj_kernel<<<4096, 256, 0, stream>>>(t_ws, wbf + 81920, gate_ws, now, nob, out);
}
